// LSTMModel_13692355740049
// MI455X (gfx1250) — matrix operand order repaired and checked by reading, not compiled, not run on hardware
//
#include <hip/hip_runtime.h>

typedef __attribute__((ext_vector_type(16))) __bf16 v16bf;
typedef __attribute__((ext_vector_type(8)))  __bf16 v8bf;
typedef __attribute__((ext_vector_type(8)))  float  v8f;

#define LSTM_T 512
#define LSTM_B 256
#define HID    256
#define GATES  1024
#define NTILES 64
#define WGS    16   // batch tiles (workgroups)
#define BT     16   // batch rows per workgroup

__device__ __forceinline__ float sigmoidf_(float x){ return 1.0f/(1.0f+__expf(-x)); }

// ---- pack layer-0 input (B,T,28) f32 -> A-fragment bf16, K padded to 32 ----
// A-frag (16-bit 16x32): lane<16: M=lane, K in {0..7,16..23}; lane>=16: M=lane-16, K in {8..15,24..31}
__global__ void pack_x0(const float* __restrict__ x, __bf16* __restrict__ out){
  int idx  = blockIdx.x*blockDim.x + threadIdx.x;   // [wg][t][lane]
  int lane = idx & 31;
  int t    = (idx >> 5) & (LSTM_T-1);
  int wg   = idx >> 14;
  if (wg >= WGS) return;
  int m = lane & 15;
  int b = wg*BT + m;
  const float* xp = x + ((size_t)b*LSTM_T + t)*28;
  __bf16* op = out + (((size_t)wg*LSTM_T + t)*32 + lane)*16;
  #pragma unroll
  for (int p = 0; p < 16; ++p){
    int k = (p & 7) + ((p >> 3) << 4) + ((lane >> 4) << 3);
    float v = (k < 28) ? xp[k] : 0.0f;
    op[p] = (__bf16)v;
  }
}

// ---- pack weight (4H x Ksrc) f32 -> B-fragment bf16 tiles [ntile][ktile][lane][16] ----
// B-frag (16-bit 32x16): lane<16: N=lane, K in {0..7,16..23}; lane>=16: N=lane-16, K in {8..15,24..31}
__global__ void pack_w(const float* __restrict__ W, __bf16* __restrict__ out,
                       int ktiles, int Ksrc){
  int idx  = blockIdx.x*blockDim.x + threadIdx.x;   // [ntile][ktile][lane]
  int lane = idx & 31;
  int kt   = (idx >> 5) % ktiles;
  int nt   = (idx >> 5) / ktiles;
  if (nt >= NTILES) return;
  int n     = lane & 15;
  int row   = nt*16 + n;
  int kbase = kt*32 + ((lane >> 4) << 3);
  __bf16* op = out + ((size_t)(nt*ktiles + kt)*32 + lane)*16;
  #pragma unroll
  for (int p = 0; p < 16; ++p){
    int k = kbase + (p & 7) + ((p >> 3) << 4);
    float v = (k < Ksrc) ? W[(size_t)row*Ksrc + k] : 0.0f;
    op[p] = (__bf16)v;
  }
}

// ---- persistent recurrent scan: one WG per 16 batch rows, full T loop ----
template<int XK>
__global__ __launch_bounds__(512, 1)
void lstm_scan(const __bf16* __restrict__ xfrag,   // [wg][T][XK][32][16]
               const __bf16* __restrict__ wihf,    // [64][XK][32][16]
               const __bf16* __restrict__ whhf,    // [64][8][32][16]
               const float*  __restrict__ bih,
               const float*  __restrict__ bhh,
               __bf16*       __restrict__ hseq){   // [wg][T][8][32][16]
  __shared__ __align__(16) float  s_gates[BT][GATES];   // 64 KB
  __shared__ __align__(32) __bf16 s_hfrag[8][32][16];   //  8 KB (h in A-frag layout)
  __shared__               float  s_c[BT][HID];         // 16 KB
  __shared__               float  s_bias[GATES];        //  4 KB

  const int tid  = threadIdx.x;
  const int lane = tid & 31;
  const int w    = tid >> 5;        // wave 0..15
  const int wg   = blockIdx.x;

  for (int i = tid; i < BT*HID;  i += 512) (&s_c[0][0])[i] = 0.0f;
  for (int i = tid; i < 8*32*16; i += 512) (&s_hfrag[0][0][0])[i] = (__bf16)0.0f;
  for (int i = tid; i < GATES;   i += 512) s_bias[i] = bih[i] + bhh[i];
  __syncthreads();

  const size_t  xstride = (size_t)XK*512;
  const __bf16* xbase   = xfrag + (size_t)wg*LSTM_T*xstride;
  __bf16*       hbase   = hseq  + (size_t)wg*LSTM_T*4096;

  const int m_ew = tid >> 5;        // row handled in elementwise phase
  const int jb   = lane * 8;        // 8 consecutive hidden cols

  for (int t = 0; t < LSTM_T; ++t){
    // Opaque zero: keeps the weight-fragment loads loop-variant so LICM cannot
    // hoist 512 VGPRs worth of B-fragments out of the t-loop (which caused
    // scratch spill/reload). Weights stream from L2 each step instead.
    int zoff;
    asm volatile("s_mov_b32 %0, 0" : "=s"(zoff));
    const __bf16* wihx = wihf + zoff;
    const __bf16* whhx = whhf + zoff;

    // ---- phase 1: gates = x_t @ Wih^T + h @ Whh^T via WMMA ----
    const __bf16* xt = xbase + (size_t)t*xstride;
    v16bf ax[XK];
    #pragma unroll
    for (int kt = 0; kt < XK; ++kt)
      ax[kt] = *(const v16bf*)(xt + ((size_t)kt*32 + lane)*16);

    #pragma unroll
    for (int nt4 = 0; nt4 < 4; ++nt4){
      const int nt = w*4 + nt4;
      v8f acc = {};
      #pragma unroll
      for (int kt = 0; kt < XK; ++kt){
        v16bf bw = *(const v16bf*)(wihx + ((size_t)(nt*XK + kt)*32 + lane)*16);
        acc = __builtin_amdgcn_wmma_f32_16x16x32_bf16(false, ax[kt], false, bw,
                                                      (short)0, acc, false, false);
      }
      #pragma unroll
      for (int kt = 0; kt < 8; ++kt){
        v16bf ah = *(const v16bf*)&s_hfrag[kt][lane][0];
        v16bf bw = *(const v16bf*)(whhx + ((size_t)(nt*8 + kt)*32 + lane)*16);
        acc = __builtin_amdgcn_wmma_f32_16x16x32_bf16(false, ah, false, bw,
                                                      (short)0, acc, false, false);
      }
      // D layout: VGPR r -> (M=r, N=lane) lanes 0-15 ; (M=r+8, N=lane-16) lanes 16-31
      const int n  = nt*16 + (lane & 15);
      const int mb = (lane >> 4) << 3;
      #pragma unroll
      for (int r = 0; r < 8; ++r) s_gates[mb + r][n] = acc[r];
    }
    __syncthreads();

    // ---- phase 2: elementwise LSTM cell update ----
    {
      const int m = m_ew;
      v8bf hv;
      #pragma unroll
      for (int q = 0; q < 8; ++q){
        const int j = jb + q;
        float ig = sigmoidf_(s_gates[m][j]       + s_bias[j]);
        float fg = sigmoidf_(s_gates[m][256 + j] + s_bias[256 + j]);
        float gg = tanhf(    s_gates[m][512 + j] + s_bias[512 + j]);
        float og = sigmoidf_(s_gates[m][768 + j] + s_bias[768 + j]);
        float c  = fg*s_c[m][j] + ig*gg;
        s_c[m][j] = c;
        hv[q] = (__bf16)(og * tanhf(c));
      }
      // write h into A-frag layout (LDS for next step, global for next layer)
      const int kt = jb >> 5;
      const int k0 = jb & 31;
      const int ln = m + (((k0 >> 3) & 1) << 4);
      const int p0 = (k0 & 7) + (((k0 >> 4) & 1) << 3);
      *(v8bf*)&s_hfrag[kt][ln][p0] = hv;
      *(v8bf*)(hbase + (size_t)t*4096 + ((size_t)kt*32 + ln)*16 + p0) = hv;
    }
    __syncthreads();
  }
}

// ---- final classifier: sigmoid(h_last @ Wout^T + bout) ----
__global__ void classifier(const __bf16* __restrict__ hseq,
                           const float*  __restrict__ Wout,
                           const float*  __restrict__ bout,
                           float*        __restrict__ out){
  const int b = blockIdx.x;
  const int o = threadIdx.x;
  if (o >= 21) return;
  const int wg = b >> 4, m = b & 15;
  const __bf16* hp = hseq + ((size_t)wg*LSTM_T + (LSTM_T-1))*4096;
  float acc = bout[o];
  for (int j = 0; j < HID; ++j){
    const int k  = j & 31, kt = j >> 5;
    const int ln = m + (((k >> 3) & 1) << 4);
    const int p  = (k & 7) + (((k >> 4) & 1) << 3);
    acc += (float)hp[((size_t)kt*32 + ln)*16 + p] * Wout[(size_t)o*HID + j];
  }
  out[(size_t)b*21 + o] = sigmoidf_(acc);
}

extern "C" void kernel_launch(void* const* d_in, const int* in_sizes, int n_in,
                              void* d_out, int out_size, void* d_ws, size_t ws_size,
                              hipStream_t stream){
  const float* x    = (const float*)d_in[0];
  const float* Wih0 = (const float*)d_in[1];
  const float* Wihr = (const float*)d_in[2];
  const float* Whh  = (const float*)d_in[3];
  const float* bih  = (const float*)d_in[4];
  const float* bhh  = (const float*)d_in[5];
  const float* Wout = (const float*)d_in[6];
  const float* bout = (const float*)d_in[7];
  float* out = (float*)d_out;

  char* ws = (char*)d_ws;
  size_t off = 0;
  auto alloc = [&](size_t bytes){
    void* p = ws + off;
    off = (off + bytes + 255) & ~(size_t)255;
    return p;
  };
  __bf16* x0f   = (__bf16*)alloc((size_t)WGS*LSTM_T*512*2);    //   8 MB
  __bf16* hA    = (__bf16*)alloc((size_t)WGS*LSTM_T*4096*2);   //  64 MB
  __bf16* hB    = (__bf16*)alloc((size_t)WGS*LSTM_T*4096*2);   //  64 MB
  __bf16* wih0f = (__bf16*)alloc((size_t)NTILES*1*512*2);
  __bf16* wihf[4]; __bf16* whhf[4];
  for (int l = 1; l < 4; ++l) wihf[l] = (__bf16*)alloc((size_t)NTILES*8*512*2);
  for (int l = 0; l < 4; ++l) whhf[l] = (__bf16*)alloc((size_t)NTILES*8*512*2);

  // pack inputs/weights into WMMA fragment layouts (bf16)
  pack_x0<<<(WGS*LSTM_T*32)/256, 256, 0, stream>>>(x, x0f);
  pack_w<<<(NTILES*1*32 + 255)/256, 256, 0, stream>>>(Wih0, wih0f, 1, 28);
  for (int l = 1; l < 4; ++l)
    pack_w<<<(NTILES*8*32)/256, 256, 0, stream>>>(Wihr + (size_t)(l-1)*GATES*HID, wihf[l], 8, 256);
  for (int l = 0; l < 4; ++l)
    pack_w<<<(NTILES*8*32)/256, 256, 0, stream>>>(Whh + (size_t)l*GATES*HID, whhf[l], 8, 256);

  // 4 recurrent layers: persistent WG per batch tile, h ping-pong in ws
  lstm_scan<1><<<WGS, 512, 0, stream>>>(x0f, wih0f,   whhf[0], bih,           bhh,           hA);
  lstm_scan<8><<<WGS, 512, 0, stream>>>(hA,  wihf[1], whhf[1], bih + GATES,   bhh + GATES,   hB);
  lstm_scan<8><<<WGS, 512, 0, stream>>>(hB,  wihf[2], whhf[2], bih + 2*GATES, bhh + 2*GATES, hA);
  lstm_scan<8><<<WGS, 512, 0, stream>>>(hA,  wihf[3], whhf[3], bih + 3*GATES, bhh + 3*GATES, hB);

  classifier<<<LSTM_B, 32, 0, stream>>>(hB, Wout, bout, out);
}
